// LSTMBaseline_30545807409199
// MI455X (gfx1250) — matrix operand order repaired and checked by reading, not compiled, not run on hardware
//
#include <hip/hip_runtime.h>

// MI455X / gfx1250, wave32. One wave per 16-row batch tile; the whole
// encoder+decoder recurrence is batch-independent, so waves never sync
// with each other. Per step per wave: 32x v_wmma_f32_16x16x32_f16.

typedef __attribute__((ext_vector_type(16))) _Float16 v16h;
typedef __attribute__((ext_vector_type(8)))  _Float16 v8h;
typedef __attribute__((ext_vector_type(8)))  float    v8f;
typedef __attribute__((ext_vector_type(4)))  float    v4f;

#define LSTM_H   64
#define TILE_M   16
#define MAX_T    512
#define BATCH    2048

__device__ __forceinline__ float fexp(float x) {
    return __builtin_amdgcn_exp2f(x * 1.44269504088896f);   // v_exp_f32
}
__device__ __forceinline__ float sigm(float x) {
    return __builtin_amdgcn_rcpf(1.0f + fexp(-x));          // v_rcp_f32
}
__device__ __forceinline__ float tanh_f(float x) {
#if __has_builtin(__builtin_amdgcn_tanhf)
    return __builtin_amdgcn_tanhf(x);                        // v_tanh_f32 (gfx1250)
#else
    return 1.0f - 2.0f * __builtin_amdgcn_rcpf(1.0f + fexp(2.0f * x));
#endif
}

extern "C" __global__ __launch_bounds__(32, 1)
void lstm_s2s_kernel(const float* __restrict__ ctx,
                     const float* __restrict__ Wih_e, const float* __restrict__ Whh_e,
                     const float* __restrict__ bih_e, const float* __restrict__ bhh_e,
                     const float* __restrict__ Wih_d, const float* __restrict__ Whh_d,
                     const float* __restrict__ bih_d, const float* __restrict__ bhh_d,
                     const float* __restrict__ Whead, const float* __restrict__ bhead,
                     float* __restrict__ out, int T, int n_steps)
{
    // B-fragments for W_hh (restaged for decoder after the encoder loop):
    //   frag f = nt*2+kt covers gate cols [16*nt,16*nt+16), K [32*kt,32*kt+32)
    //   lane L holds column N = L%16, K = 32*kt + 16*(L/16) + (0..15) (contig)
    __shared__ _Float16 sB[32][32][16];        // 32 KB
    __shared__ float    sX[MAX_T * TILE_M];    // 32 KB, transposed: sX[t*16+m]
    __shared__ _Float16 sH[TILE_M * LSTM_H];   //  2 KB, row-major h (f16)

    const int lane = threadIdx.x & 31;
    const int ln   = lane & 15;   // N (gate col) / M (row) index within tile
    const int hi   = lane >> 4;   // half-wave select
    const int b0   = blockIdx.x * TILE_M;

    // ---- one-time staging -------------------------------------------------
    auto stage_weights = [&](const float* __restrict__ Whh) {
        for (int f = 0; f < 32; ++f) {
            const int nt = f >> 1, kt = f & 1;
            const float* src = Whh + (size_t)(nt * 16 + ln) * LSTM_H + kt * 32 + 16 * hi;
            _Float16* dst = &sB[f][lane][0];
            #pragma unroll
            for (int e = 0; e < 16; ++e) dst[e] = (_Float16)src[e];
        }
    };
    stage_weights(Whh_e);

    for (int idx = lane; idx < TILE_M * T; idx += 32) {
        const int m = idx / T, t = idx - m * T;           // coalesced in t
        sX[t * TILE_M + m] = ctx[(size_t)(b0 + m) * T + t];
    }
    for (int i = lane; i < TILE_M * LSTM_H; i += 32) sH[i] = (_Float16)0.0f;

    // per-lane folded input weights / biases: n = nt*16 + ln
    float wih_e_r[16], b_e_r[16], wih_d_r[16], b_d_r[16];
    #pragma unroll
    for (int nt = 0; nt < 16; ++nt) {
        const int n = nt * 16 + ln;
        wih_e_r[nt] = Wih_e[n];
        b_e_r[nt]   = bih_e[n] + bhh_e[n];
        wih_d_r[nt] = Wih_d[n];
        b_d_r[nt]   = bih_d[n] + bhh_d[n];
    }
    float whead_r[4];
    #pragma unroll
    for (int jt = 0; jt < 4; ++jt) whead_r[jt] = Whead[jt * 16 + ln];
    const float bhead_r = bhead[0];

    v8f c_st[4];                                   // c in WMMA C/D layout
    #pragma unroll
    for (int jt = 0; jt < 4; ++jt)
        #pragma unroll
        for (int v = 0; v < 8; ++v) c_st[jt][v] = 0.0f;

    __syncthreads();   // single-wave WG: compiler/mem barrier only

    // ---- fragment loaders -------------------------------------------------
    auto load_A = [&](int kt) -> v16h {            // A: M = ln, K = 16*hi + (0..15) contig
        const int koff = kt * 32 + 16 * hi;
        const v8h lo  = *(const v8h*)(&sH[ln * LSTM_H + koff]);
        const v8h hi8 = *(const v8h*)(&sH[ln * LSTM_H + koff + 8]);
        v16h a;
        #pragma unroll
        for (int i = 0; i < 8; ++i) { a[i] = lo[i]; a[8 + i] = hi8[i]; }
        return a;
    };
    auto load_B = [&](int f) -> v16h {
        const _Float16* p = &sB[f][lane][0];
        const v8h lo  = *(const v8h*)p;
        const v8h hi8 = *(const v8h*)(p + 8);
        v16h b;
        #pragma unroll
        for (int i = 0; i < 8; ++i) { b[i] = lo[i]; b[8 + i] = hi8[i]; }
        return b;
    };

    // ---- one LSTM step. xr[v] is the scalar input for row m = v + 8*hi ----
    auto lstm_step = [&](const float* wih_r, const float* b_r,
                         const float* xr, float* zpart) {
        const v16h a0 = load_A(0);
        const v16h a1 = load_A(1);
        #pragma unroll
        for (int jt = 0; jt < 4; ++jt) {           // hidden tile j in [16jt,16jt+16)
            v8f acc[4];                            // i, f, g, o for this jt
            #pragma unroll
            for (int q = 0; q < 4; ++q) {
                const int nt = q * 4 + jt;
                v8f ini;
                #pragma unroll
                for (int v = 0; v < 8; ++v) ini[v] = xr[v] * wih_r[nt] + b_r[nt];
                const v16h bf0 = load_B(nt * 2 + 0);
                const v16h bf1 = load_B(nt * 2 + 1);
                v8f t0 = __builtin_amdgcn_wmma_f32_16x16x32_f16(
                    false, a0, false, bf0, (short)0, ini, false, false);
                acc[q] = __builtin_amdgcn_wmma_f32_16x16x32_f16(
                    false, a1, false, bf1, (short)0, t0, false, false);
            }
            #pragma unroll
            for (int v = 0; v < 8; ++v) {          // element (m = v+8*hi, j = 16jt+ln)
                const float ig = sigm(acc[0][v]);
                const float fg = sigm(acc[1][v]);
                const float gg = tanh_f(acc[2][v]);
                const float og = sigm(acc[3][v]);
                const float cn = fg * c_st[jt][v] + ig * gg;
                c_st[jt][v] = cn;
                const float hn = og * tanh_f(cn);
                sH[(v + 8 * hi) * LSTM_H + jt * 16 + ln] = (_Float16)hn;
                if (zpart) zpart[v] += hn * whead_r[jt];
            }
        }
    };

    // ---- encoder ----------------------------------------------------------
    for (int t = 0; t < T; ++t) {
        const v4f xa = *(const v4f*)(&sX[t * TILE_M + 8 * hi]);
        const v4f xb = *(const v4f*)(&sX[t * TILE_M + 8 * hi + 4]);
        const float xr[8] = {xa[0], xa[1], xa[2], xa[3], xb[0], xb[1], xb[2], xb[3]};
        lstm_step(wih_e_r, b_e_r, xr, nullptr);
        __syncthreads();
    }

    // ---- restage decoder weights over encoder weights ---------------------
    stage_weights(Whh_d);
    __syncthreads();

    // ---- decoder (autoregressive) -----------------------------------------
    float zr[8];
    {
        const v4f za = *(const v4f*)(&sX[(T - 1) * TILE_M + 8 * hi]);
        const v4f zb = *(const v4f*)(&sX[(T - 1) * TILE_M + 8 * hi + 4]);
        #pragma unroll
        for (int v = 0; v < 4; ++v) { zr[v] = za[v]; zr[4 + v] = zb[v]; }
    }
    for (int s = 0; s < n_steps; ++s) {
        float zp[8] = {0, 0, 0, 0, 0, 0, 0, 0};
        lstm_step(wih_d_r, b_d_r, zr, zp);
        #pragma unroll
        for (int v = 0; v < 8; ++v) {              // reduce over j within 16-lane half
            float sum = zp[v];
            sum += __shfl_xor(sum, 1, 32);
            sum += __shfl_xor(sum, 2, 32);
            sum += __shfl_xor(sum, 4, 32);
            sum += __shfl_xor(sum, 8, 32);
            zr[v] = sum + bhead_r;                 // z for row m = v + 8*hi
        }
        #pragma unroll
        for (int v = 0; v < 8; ++v)
            if (ln == v)
                out[(size_t)(b0 + 8 * hi + v) * n_steps + s] = zr[v];
        __syncthreads();
    }
}

extern "C" void kernel_launch(void* const* d_in, const int* in_sizes, int n_in,
                              void* d_out, int out_size, void* d_ws, size_t ws_size,
                              hipStream_t stream) {
    (void)n_in; (void)d_ws; (void)ws_size;
    const float* ctx   = (const float*)d_in[0];
    const float* Wih_e = (const float*)d_in[1];
    const float* Whh_e = (const float*)d_in[2];
    const float* bih_e = (const float*)d_in[3];
    const float* bhh_e = (const float*)d_in[4];
    const float* Wih_d = (const float*)d_in[5];
    const float* Whh_d = (const float*)d_in[6];
    const float* bih_d = (const float*)d_in[7];
    const float* bhh_d = (const float*)d_in[8];
    const float* Whead = (const float*)d_in[9];
    const float* bhead = (const float*)d_in[10];
    // d_in[11] is n_steps on device; host-side we recover it from out_size.
    const int B = BATCH;                 // per reference setup_inputs
    const int T = in_sizes[0] / B;       // 512
    const int n_steps = out_size / B;    // 256
    dim3 grid(B / TILE_M), block(32);
    hipLaunchKernelGGL(lstm_s2s_kernel, grid, block, 0, stream,
                       ctx, Wih_e, Whh_e, bih_e, bhh_e,
                       Wih_d, Whh_d, bih_d, bhh_d,
                       Whead, bhead, (float*)d_out, T, n_steps);
}
